// SelfAttention_3728031613773
// MI455X (gfx1250) — matrix operand order repaired and checked by reading, not compiled, not run on hardware
//
#include <hip/hip_runtime.h>
#include <hip/hip_bf16.h>
#include <math.h>

// ---------------------------------------------------------------------------
// Problem constants (match the reference)
// ---------------------------------------------------------------------------
#define BATCH   2
#define SEQ     2048
#define DMODEL  2048
#define NHEADS  16
#define DHEAD   128
#define THREED  6144
#define MTOT    (BATCH * SEQ)          // 4096 rows for the two dense GEMMs

typedef __attribute__((ext_vector_type(16))) __bf16 v16bf;
typedef __attribute__((ext_vector_type(8)))  float  v8f;
typedef __attribute__((ext_vector_type(4)))  unsigned int u32x4;
typedef __attribute__((ext_vector_type(8)))  unsigned int u32x8;

union FragB { unsigned int u[8]; v16bf v; };   // 16 bf16 operand fragment
union FragA { float f[8];        v8f  v; };    // 8 f32 accumulator fragment

__device__ __forceinline__ unsigned short f2bf(float x) {
    unsigned int u = __float_as_uint(x);
    u = (u + 0x7FFFu + ((u >> 16) & 1u)) >> 16;   // round-to-nearest-even
    return (unsigned short)u;
}

// ---------------------------------------------------------------------------
// Tensor Data Mover: DMA a 2-D bf16 tile (rows x cols, row stride in elems)
// from global memory into LDS. D# layout per CDNA5 ISA ch.8:
//   group0: [1:0]=count, [63:32]=lds_addr, [120:64]=global_addr, [127:126]=2
//   group1: [17:16]=data_size(1->2B), [79:48]=tensor_dim0, [111:80]=tensor_dim1,
//           [127:112]=tile_dim0, [143:128]=tile_dim1, [207:160]=dim0_stride
// Issued once per wave (EXEC ignored); tracked by TENSORcnt.
// ---------------------------------------------------------------------------
__device__ __forceinline__ void tdm_load_tile_2d(unsigned lds_addr,
                                                 const unsigned short* gptr,
                                                 unsigned rows, unsigned cols,
                                                 unsigned rowStrideElems) {
    unsigned long long ga = (unsigned long long)(uintptr_t)gptr;
    u32x4 g0;
    g0[0] = 1u;                                            // count=1, user mode
    g0[1] = lds_addr;                                      // LDS byte address
    g0[2] = (unsigned)ga;                                  // global_addr[31:0]
    g0[3] = (unsigned)((ga >> 32) & 0x01FFFFFFu)           // global_addr[56:32]
          | (2u << 30);                                    // type = 2 (image)
    u32x8 g1;
    g1[0] = 1u << 16;                                      // data_size=1 (2 bytes)
    g1[1] = cols << 16;                                    // tensor_dim0[15:0]
    g1[2] = rows << 16;                                    // tensor_dim1[15:0]
    g1[3] = cols << 16;                                    // tile_dim0
    g1[4] = rows;                                          // tile_dim1 (tile_dim2=0)
    g1[5] = rowStrideElems;                                // tensor_dim0_stride[31:0]
    g1[6] = 0u;
    g1[7] = 0u;
    asm volatile("tensor_load_to_lds %0, %1" :: "s"(g0), "s"(g1) : "memory");
}

// ---------------------------------------------------------------------------
// fp32 -> bf16 conversion
// ---------------------------------------------------------------------------
__global__ __launch_bounds__(256)
void cvt_bf16(const float* __restrict__ in, unsigned short* __restrict__ out, int n) {
    int i = blockIdx.x * 256 + threadIdx.x;
    if (i < n) out[i] = f2bf(in[i]);
}

// ---------------------------------------------------------------------------
// QKV projection: C[4096,6144] = Xbf[4096,2048] @ Wbf[2048,6144] + bias
// Epilogue scatters into head-major Q/K/V [B][H][S][DHEAD] bf16 buffers.
// Block tile 128x64, BK=32, 256 threads (8 waves), wave tile 32x32.
// ---------------------------------------------------------------------------
__global__ __launch_bounds__(256)
void qkv_gemm(const unsigned short* __restrict__ A,    // [4096][2048] bf16
              const unsigned short* __restrict__ Bm,   // [2048][6144] bf16
              const float* __restrict__ bias,          // [6144] f32
              unsigned short* __restrict__ Qo,
              unsigned short* __restrict__ Ko,
              unsigned short* __restrict__ Vo) {
    __shared__ __attribute__((aligned(16))) unsigned short As[128][32];  // 8KB
    __shared__ __attribute__((aligned(16))) unsigned short Bs[32][64];   // 4KB

    const int KDIM = DMODEL, NDIM = THREED;
    const int m0 = blockIdx.y * 128;
    const int n0 = blockIdx.x * 64;
    const int tid  = threadIdx.x;
    const int lane = tid & 31;
    const int wid  = tid >> 5;
    const int wm = (wid & 3) * 32;     // wave m offset in block tile
    const int wn = (wid >> 2) * 32;    // wave n offset in block tile
    const int lr = lane & 15, lh = lane >> 4;

    FragA acc[2][2];
#pragma unroll
    for (int mi = 0; mi < 2; ++mi)
#pragma unroll
        for (int ni = 0; ni < 2; ++ni)
#pragma unroll
            for (int r = 0; r < 8; ++r) acc[mi][ni].f[r] = 0.0f;

    for (int k0 = 0; k0 < KDIM; k0 += 32) {
        // cooperative A tile load: 256 thr * 16 bf16 = 128x32
        {
            int row = tid >> 1;
            int col = (tid & 1) * 16;
            const uint4* src = (const uint4*)(A + (size_t)(m0 + row) * KDIM + k0 + col);
            uint4* dst = (uint4*)(&As[row][col]);
            dst[0] = src[0];
            dst[1] = src[1];
        }
        // cooperative B tile load: 256 thr * 8 bf16 = 32x64
        {
            int row = tid >> 3;
            int col = (tid & 7) * 8;
            *(uint4*)(&Bs[row][col]) =
                *(const uint4*)(Bm + (size_t)(k0 + row) * NDIM + n0 + col);
        }
        __syncthreads();

        FragB af[2], bf[2];
#pragma unroll
        for (int mi = 0; mi < 2; ++mi) {
            const unsigned short* ap = &As[wm + mi * 16 + lr][0];
#pragma unroll
            for (int r = 0; r < 4; ++r)
                af[mi].u[r] = *(const unsigned int*)(ap + lh * 8 + 2 * r);
#pragma unroll
            for (int r = 0; r < 4; ++r)
                af[mi].u[4 + r] = *(const unsigned int*)(ap + 16 + lh * 8 + 2 * r);
        }
#pragma unroll
        for (int ni = 0; ni < 2; ++ni) {
            const unsigned short* bp = &Bs[0][wn + ni * 16 + lr];
#pragma unroll
            for (int r = 0; r < 4; ++r) {
                bf[ni].u[r] = (unsigned int)bp[(lh * 8 + 2 * r) * 64] | ((unsigned int)bp[(lh * 8 + 2 * r + 1) * 64] << 16);
                bf[ni].u[4 + r] = (unsigned int)bp[(16 + lh * 8 + 2 * r) * 64] | ((unsigned int)bp[(17 + lh * 8 + 2 * r) * 64] << 16);
            }
        }
#pragma unroll
        for (int mi = 0; mi < 2; ++mi)
#pragma unroll
            for (int ni = 0; ni < 2; ++ni)
                acc[mi][ni].v = __builtin_amdgcn_wmma_f32_16x16x32_bf16(
                    false, af[mi].v, false, bf[ni].v, (short)0, acc[mi][ni].v, false, false);
        __syncthreads();
    }

    // epilogue: bias add, convert, scatter to Q/K/V head-major
#pragma unroll
    for (int mi = 0; mi < 2; ++mi)
#pragma unroll
        for (int ni = 0; ni < 2; ++ni)
#pragma unroll
            for (int r = 0; r < 8; ++r) {
                int m = m0 + wm + mi * 16 + r + 8 * lh;
                int n = n0 + wn + ni * 16 + lr;
                float val = acc[mi][ni].f[r] + bias[n];
                unsigned short bv = f2bf(val);
                int b = m >> 11, s = m & (SEQ - 1);
                int part = n >> 11;          // 0=Q 1=K 2=V
                int nn = n & (DMODEL - 1);
                int head = nn >> 7, d = nn & (DHEAD - 1);
                size_t off = ((((size_t)b * NHEADS + head) * SEQ) + s) * DHEAD + d;
                unsigned short* dst = (part == 0) ? Qo : ((part == 1) ? Ko : Vo);
                dst[off] = bv;
            }
}

// ---------------------------------------------------------------------------
// Causal flash attention. One block = (q-tile of 64, head, batch).
// 128 threads = 4 waves; each wave owns 16 query rows.
// K/V tiles of 32 keys DMA'd into LDS by the Tensor Data Mover (wave 0),
// shared by all waves.
// ---------------------------------------------------------------------------
__global__ __launch_bounds__(128)
void flash_attn(const unsigned short* __restrict__ Qg,
                const unsigned short* __restrict__ Kg,
                const unsigned short* __restrict__ Vg,
                unsigned short* __restrict__ Og)    // [B][S][DMODEL] bf16
{
    __shared__ __attribute__((aligned(16))) unsigned short Kt[32][DHEAD];   // 8KB
    __shared__ __attribute__((aligned(16))) unsigned short Vt[32][DHEAD];   // 8KB
    __shared__ __attribute__((aligned(16))) unsigned short Pt[4][16][32];   // 4KB

    const int qt = blockIdx.x, h = blockIdx.y, b = blockIdx.z;
    const int qb = qt * 64;
    const int tid = threadIdx.x, lane = tid & 31, w = tid >> 5;
    const int lr = lane & 15, lh = lane >> 4;

    const size_t headOff = (((size_t)b * NHEADS + h) * SEQ) * DHEAD;
    const unsigned short* Qh = Qg + headOff;
    const unsigned short* Kh = Kg + headOff;
    const unsigned short* Vh = Vg + headOff;

    const unsigned ldsK = (unsigned)(uintptr_t)&Kt[0][0];
    const unsigned ldsV = (unsigned)(uintptr_t)&Vt[0][0];

    // Q fragments for this wave's 16 rows: 4 chunks of 32 dims (A layout)
    FragB qf[4];
    {
        const int qrow = qb + w * 16 + lr;
        const unsigned short* qp = Qh + (size_t)qrow * DHEAD;
#pragma unroll
        for (int c = 0; c < 4; ++c) {
#pragma unroll
            for (int r = 0; r < 4; ++r)
                qf[c].u[r] = *(const unsigned int*)(qp + c * 32 + lh * 8 + 2 * r);
#pragma unroll
            for (int r = 0; r < 4; ++r)
                qf[c].u[4 + r] = *(const unsigned int*)(qp + c * 32 + 16 + lh * 8 + 2 * r);
        }
    }

    FragA of[8];
    float mst[8], lst[8];
#pragma unroll
    for (int j = 0; j < 8; ++j)
#pragma unroll
        for (int r = 0; r < 8; ++r) of[j].f[r] = 0.0f;
#pragma unroll
    for (int r = 0; r < 8; ++r) { mst[r] = -INFINITY; lst[r] = 0.0f; }

    const float scale = 0.08838834764831845f;   // 1/sqrt(128)
    const int ktmax = (qb + 64) / 32;           // causal: keys < qb+64

    for (int kt = 0; kt < ktmax; ++kt) {
        const int kb = kt * 32;

        // K/V tiles (32 x 128 bf16 each) via Tensor Data Mover, wave 0 only
        if (w == 0) {
            tdm_load_tile_2d(ldsK, Kh + (size_t)kb * DHEAD, 32u, (unsigned)DHEAD, (unsigned)DHEAD);
            tdm_load_tile_2d(ldsV, Vh + (size_t)kb * DHEAD, 32u, (unsigned)DHEAD, (unsigned)DHEAD);
            __builtin_amdgcn_s_wait_tensorcnt(0);
        } else if (w == 1 && kt + 1 < ktmax) {
            // warm L2 for the next tile while the DMA runs
            __builtin_prefetch(Kh + (size_t)(kb + 32 + lane) * DHEAD, 0, 1);
            __builtin_prefetch(Vh + (size_t)(kb + 32 + lane) * DHEAD, 0, 1);
        }
        __syncthreads();

        // S = Q @ K^T  (two 16-col tiles, contraction over 4 chunks of 32 dims)
        FragA s[2];
#pragma unroll
        for (int nt = 0; nt < 2; ++nt) {
#pragma unroll
            for (int r = 0; r < 8; ++r) s[nt].f[r] = 0.0f;
#pragma unroll
            for (int c = 0; c < 4; ++c) {
                FragB kf;   // B frag: lane%16 = key index, elements = dim (table order)
#pragma unroll
                for (int r = 0; r < 4; ++r) {
                    kf.u[r] = *(const unsigned int*)(&Kt[nt * 16 + lr][c * 32 + lh * 8 + 2 * r]);
                    kf.u[4 + r] = *(const unsigned int*)(&Kt[nt * 16 + lr][c * 32 + 16 + lh * 8 + 2 * r]);
                }
                s[nt].v = __builtin_amdgcn_wmma_f32_16x16x32_bf16(
                    false, qf[c].v, false, kf.v, (short)0, s[nt].v, false, false);
            }
        }

        // online softmax (row = r + 8*lh within wave tile, col = lr)
        const int myq = qb + w * 16;
        float pv0[8], pv1[8];
#pragma unroll
        for (int r = 0; r < 8; ++r) {
            int row = myq + r + 8 * lh;
            float x0 = s[0].f[r], x1 = s[1].f[r];
            x0 = (kb + lr      <= row) ? x0 * scale : -INFINITY;
            x1 = (kb + 16 + lr <= row) ? x1 * scale : -INFINITY;
            float tm = fmaxf(x0, x1);
#pragma unroll
            for (int d = 1; d < 16; d <<= 1) tm = fmaxf(tm, __shfl_xor(tm, d, 32));
            float mnew = fmaxf(mst[r], tm);
            float corr = __expf(mst[r] - mnew);
            float p0 = __expf(x0 - mnew);
            float p1 = __expf(x1 - mnew);
            float rs = p0 + p1;
#pragma unroll
            for (int d = 1; d < 16; d <<= 1) rs += __shfl_xor(rs, d, 32);
            lst[r] = lst[r] * corr + rs;
            mst[r] = mnew;
            pv0[r] = p0; pv1[r] = p1;
#pragma unroll
            for (int j = 0; j < 8; ++j) of[j].f[r] *= corr;
        }

        // relayout P (D layout) -> A layout via LDS
#pragma unroll
        for (int r = 0; r < 8; ++r) {
            Pt[w][r + 8 * lh][lr]      = f2bf(pv0[r]);
            Pt[w][r + 8 * lh][16 + lr] = f2bf(pv1[r]);
        }
        __syncthreads();

        FragB pf;
        {
            const unsigned short* pp = &Pt[w][lr][0];
#pragma unroll
            for (int r = 0; r < 4; ++r)
                pf.u[r] = *(const unsigned int*)(pp + lh * 8 + 2 * r);
#pragma unroll
            for (int r = 0; r < 4; ++r)
                pf.u[4 + r] = *(const unsigned int*)(pp + 16 + lh * 8 + 2 * r);
        }

        // O += P @ V  (8 dh chunks of 16 cols; V B-frag gathered down Vt rows, stride DHEAD)
#pragma unroll
        for (int j = 0; j < 8; ++j) {
            FragB vf;
            const unsigned short* vp = &Vt[0][j * 16 + lr];
#pragma unroll
            for (int r = 0; r < 4; ++r) {
                vf.u[r] = (unsigned int)vp[(lh * 8 + 2 * r) * DHEAD] | ((unsigned int)vp[(lh * 8 + 2 * r + 1) * DHEAD] << 16);
                vf.u[4 + r] = (unsigned int)vp[(16 + lh * 8 + 2 * r) * DHEAD] | ((unsigned int)vp[(17 + lh * 8 + 2 * r) * DHEAD] << 16);
            }
            of[j].v = __builtin_amdgcn_wmma_f32_16x16x32_bf16(
                false, pf.v, false, vf.v, (short)0, of[j].v, false, false);
        }
        __syncthreads();
    }

    // normalize and store attention output bf16 [B][S][DMODEL]
    unsigned short* Ob = Og + ((size_t)b * SEQ + qb + w * 16) * DMODEL + h * DHEAD;
#pragma unroll
    for (int j = 0; j < 8; ++j)
#pragma unroll
        for (int r = 0; r < 8; ++r) {
            int m = r + 8 * lh;
            float val = of[j].f[r] / lst[r];
            Ob[(size_t)m * DMODEL + j * 16 + lr] = f2bf(val);
        }
}

// ---------------------------------------------------------------------------
// Output projection: out[4096,2048] = Attn_bf[4096,2048] @ Wout_bf[2048,2048] + bias
// fp32 output. Same tiling as qkv_gemm.
// ---------------------------------------------------------------------------
__global__ __launch_bounds__(256)
void out_gemm(const unsigned short* __restrict__ A,    // [4096][2048] bf16
              const unsigned short* __restrict__ Bm,   // [2048][2048] bf16
              const float* __restrict__ bias,          // [2048] f32
              float* __restrict__ Co) {                // [4096][2048] f32
    __shared__ __attribute__((aligned(16))) unsigned short As[128][32];
    __shared__ __attribute__((aligned(16))) unsigned short Bs[32][64];

    const int KDIM = DMODEL, NDIM = DMODEL;
    const int m0 = blockIdx.y * 128;
    const int n0 = blockIdx.x * 64;
    const int tid  = threadIdx.x;
    const int lane = tid & 31;
    const int wid  = tid >> 5;
    const int wm = (wid & 3) * 32;
    const int wn = (wid >> 2) * 32;
    const int lr = lane & 15, lh = lane >> 4;

    FragA acc[2][2];
#pragma unroll
    for (int mi = 0; mi < 2; ++mi)
#pragma unroll
        for (int ni = 0; ni < 2; ++ni)
#pragma unroll
            for (int r = 0; r < 8; ++r) acc[mi][ni].f[r] = 0.0f;

    for (int k0 = 0; k0 < KDIM; k0 += 32) {
        {
            int row = tid >> 1;
            int col = (tid & 1) * 16;
            const uint4* src = (const uint4*)(A + (size_t)(m0 + row) * KDIM + k0 + col);
            uint4* dst = (uint4*)(&As[row][col]);
            dst[0] = src[0];
            dst[1] = src[1];
        }
        {
            int row = tid >> 3;
            int col = (tid & 7) * 8;
            *(uint4*)(&Bs[row][col]) =
                *(const uint4*)(Bm + (size_t)(k0 + row) * NDIM + n0 + col);
        }
        __syncthreads();

        FragB af[2], bf[2];
#pragma unroll
        for (int mi = 0; mi < 2; ++mi) {
            const unsigned short* ap = &As[wm + mi * 16 + lr][0];
#pragma unroll
            for (int r = 0; r < 4; ++r)
                af[mi].u[r] = *(const unsigned int*)(ap + lh * 8 + 2 * r);
#pragma unroll
            for (int r = 0; r < 4; ++r)
                af[mi].u[4 + r] = *(const unsigned int*)(ap + 16 + lh * 8 + 2 * r);
        }
#pragma unroll
        for (int ni = 0; ni < 2; ++ni) {
            const unsigned short* bp = &Bs[0][wn + ni * 16 + lr];
#pragma unroll
            for (int r = 0; r < 4; ++r) {
                bf[ni].u[r] = (unsigned int)bp[(lh * 8 + 2 * r) * 64] | ((unsigned int)bp[(lh * 8 + 2 * r + 1) * 64] << 16);
                bf[ni].u[4 + r] = (unsigned int)bp[(16 + lh * 8 + 2 * r) * 64] | ((unsigned int)bp[(17 + lh * 8 + 2 * r) * 64] << 16);
            }
        }
#pragma unroll
        for (int mi = 0; mi < 2; ++mi)
#pragma unroll
            for (int ni = 0; ni < 2; ++ni)
                acc[mi][ni].v = __builtin_amdgcn_wmma_f32_16x16x32_bf16(
                    false, af[mi].v, false, bf[ni].v, (short)0, acc[mi][ni].v, false, false);
        __syncthreads();
    }

#pragma unroll
    for (int mi = 0; mi < 2; ++mi)
#pragma unroll
        for (int ni = 0; ni < 2; ++ni)
#pragma unroll
            for (int r = 0; r < 8; ++r) {
                int m = m0 + wm + mi * 16 + r + 8 * lh;
                int n = n0 + wn + ni * 16 + lr;
                Co[(size_t)m * NDIM + n] = acc[mi][ni].f[r] + bias[n];
            }
}

// ---------------------------------------------------------------------------
// Host entry
// ---------------------------------------------------------------------------
extern "C" void kernel_launch(void* const* d_in, const int* in_sizes, int n_in,
                              void* d_out, int out_size, void* d_ws, size_t ws_size,
                              hipStream_t stream) {
    (void)in_sizes; (void)n_in; (void)out_size; (void)ws_size;

    const float* x     = (const float*)d_in[0];   // [2,2048,2048]
    const float* w_in  = (const float*)d_in[1];   // [2048,6144]
    const float* b_in  = (const float*)d_in[2];   // [6144]
    const float* w_out = (const float*)d_in[3];   // [2048,2048]
    const float* b_out = (const float*)d_in[4];   // [2048]
    float* out = (float*)d_out;                   // [2,2048,2048] f32

    const size_t MB = 1u << 20;
    char* ws = (char*)d_ws;
    unsigned short* xbf    = (unsigned short*)(ws + 0);          // 16MB (reused as attn later)
    unsigned short* attn   = (unsigned short*)(ws + 0);          // alias of xbf (x dead after QKV)
    unsigned short* winbf  = (unsigned short*)(ws + 16 * MB);    // 24MB
    unsigned short* woutbf = (unsigned short*)(ws + 40 * MB);    // 8MB
    unsigned short* qbuf   = (unsigned short*)(ws + 48 * MB);    // 16MB
    unsigned short* kbuf   = (unsigned short*)(ws + 64 * MB);    // 16MB
    unsigned short* vbuf   = (unsigned short*)(ws + 80 * MB);    // 16MB  -> 96MB total

    const int nX = MTOT * DMODEL;          // 8,388,608
    const int nWi = DMODEL * THREED;       // 12,582,912
    const int nWo = DMODEL * DMODEL;       // 4,194,304

    cvt_bf16<<<(nX + 255) / 256, 256, 0, stream>>>(x, xbf, nX);
    cvt_bf16<<<(nWi + 255) / 256, 256, 0, stream>>>(w_in, winbf, nWi);
    cvt_bf16<<<(nWo + 255) / 256, 256, 0, stream>>>(w_out, woutbf, nWo);

    qkv_gemm<<<dim3(THREED / 64, MTOT / 128), 256, 0, stream>>>(
        xbf, winbf, b_in, qbuf, kbuf, vbuf);

    flash_attn<<<dim3(SEQ / 64, NHEADS, BATCH), 128, 0, stream>>>(
        qbuf, kbuf, vbuf, attn);

    out_gemm<<<dim3(DMODEL / 64, MTOT / 128), 256, 0, stream>>>(
        attn, woutbf, b_out, out);
}
